// VoxelGrid_11184094839333
// MI455X (gfx1250) — matrix operand order repaired and checked by reading, not compiled, not run on hardware
//
#include <hip/hip_runtime.h>
#include <hip/hip_bf16.h>
#include <stdint.h>

#define NUM_LEVELS 16
#define LOG2_HASH  19
#define HASH_MASK  ((1u << LOG2_HASH) - 1u)
#define NPTS       262144
#define DENSITY_N  (128 * 128 * 128)

typedef __attribute__((ext_vector_type(16))) _Float16 v16h;
typedef __attribute__((ext_vector_type(8)))  _Float16 v8h;
typedef __attribute__((ext_vector_type(8)))  float    v8f;
typedef __attribute__((ext_vector_type(4)))  float    v4f;
typedef __attribute__((ext_vector_type(4)))  int      v4i;

__device__ __forceinline__ float sigmoidf_dev(float x) {
    return 1.0f / (1.0f + __expf(-x));
}

// ---------------------------------------------------------------------------
// Kernel 1: density = sigmoid(density_grid), vectorized float4
// ---------------------------------------------------------------------------
__global__ void density_sigmoid_kernel(const float* __restrict__ g,
                                       float* __restrict__ out) {
    int i = blockIdx.x * blockDim.x + threadIdx.x;   // one float4 per thread
    const v4f* gi = (const v4f*)g;
    v4f x = gi[i];
    v4f r;
#pragma unroll
    for (int j = 0; j < 4; ++j) r[j] = sigmoidf_dev(x[j]);
    ((v4f*)out)[i] = r;
}

// ---------------------------------------------------------------------------
// Kernel 2: pre-pack W1/W2/W3 (f32, row-major) into f16 B-matrix lane layout.
// B 32x16 f16 layout (wave32): lane l holds column n = l&15,
//   rows K = (l>>4)*8 + j (j<8), 16 + (l>>4)*8 + (j-8) (j>=8), element j -> VGPR j/2 half j%2.
// Tile order: W1 tiles t=0..63 (t = kt*4+nt), W2 t=64..71, W3 t=72..73
// (W3 cols >= 3 zero-padded). (tile,lane) owns 16 halfs at pk + (t*32+lane)*16.
// ---------------------------------------------------------------------------
__global__ void pack_weights_kernel(const float* __restrict__ W1,
                                    const float* __restrict__ W2,
                                    const float* __restrict__ W3,
                                    _Float16* __restrict__ pk) {
    int tid = blockIdx.x * blockDim.x + threadIdx.x;
    if (tid >= 74 * 32) return;
    int t    = tid >> 5;
    int lane = tid & 31;
    int krow = (lane >> 4) << 3;     // 0 or 8 (K within the 32-tile)
    int nlo  = lane & 15;
    _Float16* dst = pk + (size_t)tid * 16;

    if (t < 64) {                                    // W1 [512,64]
        int kt = t >> 2, nt = t & 3;
        const float* src = W1 + (size_t)(kt * 32 + krow) * 64 + (nt * 16 + nlo);
#pragma unroll
        for (int j = 0; j < 16; ++j) dst[j] = (_Float16)src[(j + (j & 8)) * 64];
    } else if (t < 72) {                             // W2 [64,64]
        int tt = t - 64;
        int kt = tt >> 2, nt = tt & 3;
        const float* src = W2 + (size_t)(kt * 32 + krow) * 64 + (nt * 16 + nlo);
#pragma unroll
        for (int j = 0; j < 16; ++j) dst[j] = (_Float16)src[(j + (j & 8)) * 64];
    } else {                                         // W3 [64,3] zero-padded to 16
        int kt = t - 72;
        const float* src = W3 + (size_t)(kt * 32 + krow) * 3 + nlo;
#pragma unroll
        for (int j = 0; j < 16; ++j)
            dst[j] = (nlo < 3) ? (_Float16)src[(j + (j & 8)) * 3] : (_Float16)0.0f;
    }
}

// ---------------------------------------------------------------------------
// Kernel 3: fused hash-gather + 3-layer MLP.
// Block = 256 threads = 8 waves; each wave owns one 16-point M-tile.
// LDS halfs layout: [0,32768) W1pack | [32768,36864) W2pack |
//                   [36864,37888) W3pack | [37888,46080) per-wave H (16x64)
// ---------------------------------------------------------------------------
__global__ void __launch_bounds__(256)
ngp_mlp_kernel(const float* __restrict__ xyz,
               const float* __restrict__ grids,
               const _Float16* __restrict__ pk,
               const float* __restrict__ b1,
               const float* __restrict__ b2,
               const float* __restrict__ b3,
               float* __restrict__ color) {
    __shared__ __align__(16) _Float16 smem[46080];   // 92160 B

    const int tid = threadIdx.x;
    // cooperative stage of packed weights: 37888 halfs = 4736 x 16B
    {
        const v4i* src = (const v4i*)pk;
        v4i* dst = (v4i*)smem;
        for (int i = tid; i < 4736; i += 256) dst[i] = src[i];
    }
    __syncthreads();

    const int wave = tid >> 5;
    const int lane = tid & 31;
    const int m    = lane & 15;          // row within M-tile this lane feeds (A)
    const int kb   = (lane >> 4) << 3;   // A K-base: 0 or 8
    const int n0   = lane & 15;          // column this lane holds (B/C/D)
    const int mrow = (lane >> 4) << 3;   // C/D row base: 0 or 8
    const int base = (blockIdx.x * 8 + wave) * 16;

    const _Float16* sW1 = smem;
    const _Float16* sW2 = smem + 32768;
    const _Float16* sW3 = smem + 36864;
    _Float16*       sH  = smem + 37888 + wave * 1024;

    const float px = xyz[(base + m) * 3 + 0];
    const float py = xyz[(base + m) * 3 + 1];
    const float pz = xyz[(base + m) * 3 + 2];

    // uniform 64-bit base + 32-bit per-lane byte offset (max 2^30 < 2^31):
    // lets the backend use saddr-form global_load_b128 with imm offsets.
    const char* gbase = (const char*)grids;
    const uint32_t kboff = (uint32_t)kb * 4u;

    // ---------------- layer 1: 16 levels == 16 K-tiles of 32 ----------------
    v8f acc[4] = {v8f{}, v8f{}, v8f{}, v8f{}};
    for (int L = 0; L < NUM_LEVELS; ++L) {
        float rf = (float)(16u << L);
        uint32_t ix = (uint32_t)(px * rf);
        uint32_t iy = (uint32_t)(py * rf);
        uint32_t iz = (uint32_t)(pz * rf);
        uint32_t h  = (ix ^ (iy * 2654435761u) ^ (iz * 805459861u)) & HASH_MASK;
        // 128 bytes per hash entry; level stride = 2^19 * 128 = 2^26 bytes
        uint32_t off = (((uint32_t)L << LOG2_HASH) + h) * 128u + kboff;

        // lane grabs exactly its A-slots: K = kb+{0..7} and 16+kb+{0..7}
        v4f f0 = *(const v4f*)(gbase + off);
        v4f f1 = *(const v4f*)(gbase + off + 16);
        v4f f2 = *(const v4f*)(gbase + off + 64);
        v4f f3 = *(const v4f*)(gbase + off + 80);
        v16h a;
#pragma unroll
        for (int j = 0; j < 4; ++j) {
            a[j]      = (_Float16)f0[j];
            a[j + 4]  = (_Float16)f1[j];
            a[j + 8]  = (_Float16)f2[j];
            a[j + 12] = (_Float16)f3[j];
        }
#pragma unroll
        for (int nt = 0; nt < 4; ++nt) {
            v16h b = *(const v16h*)(sW1 + ((L * 4 + nt) * 32 + lane) * 16);
            acc[nt] = __builtin_amdgcn_wmma_f32_16x16x32_f16(
                false, a, false, b, (short)0, acc[nt], false, false);
        }
    }

    // bias + relu, store H1 to LDS as f16 (row-major 16x64)
#pragma unroll
    for (int nt = 0; nt < 4; ++nt) {
        float bb = b1[nt * 16 + n0];
#pragma unroll
        for (int v = 0; v < 8; ++v)
            sH[(mrow + v) * 64 + nt * 16 + n0] =
                (_Float16)fmaxf(acc[nt][v] + bb, 0.0f);
    }
    __syncthreads();

    // ---------------- layer 2: K=64 (2 K-tiles) ----------------
    v8f acc2[4] = {v8f{}, v8f{}, v8f{}, v8f{}};
#pragma unroll
    for (int kt = 0; kt < 2; ++kt) {
        const _Float16* hp = sH + m * 64 + kt * 32 + kb;
        v8h lo = *(const v8h*)hp;
        v8h hi = *(const v8h*)(hp + 16);
        v16h a;
#pragma unroll
        for (int j = 0; j < 8; ++j) { a[j] = lo[j]; a[j + 8] = hi[j]; }
#pragma unroll
        for (int nt = 0; nt < 4; ++nt) {
            v16h b = *(const v16h*)(sW2 + ((kt * 4 + nt) * 32 + lane) * 16);
            acc2[nt] = __builtin_amdgcn_wmma_f32_16x16x32_f16(
                false, a, false, b, (short)0, acc2[nt], false, false);
        }
    }
    __syncthreads();

    // bias + relu, overwrite sH with H2
#pragma unroll
    for (int nt = 0; nt < 4; ++nt) {
        float bb = b2[nt * 16 + n0];
#pragma unroll
        for (int v = 0; v < 8; ++v)
            sH[(mrow + v) * 64 + nt * 16 + n0] =
                (_Float16)fmaxf(acc2[nt][v] + bb, 0.0f);
    }
    __syncthreads();

    // ---------------- layer 3: K=64 -> 16 cols (3 valid) ----------------
    v8f acc3 = v8f{};
#pragma unroll
    for (int kt = 0; kt < 2; ++kt) {
        const _Float16* hp = sH + m * 64 + kt * 32 + kb;
        v8h lo = *(const v8h*)hp;
        v8h hi = *(const v8h*)(hp + 16);
        v16h a;
#pragma unroll
        for (int j = 0; j < 8; ++j) { a[j] = lo[j]; a[j + 8] = hi[j]; }
        v16h b = *(const v16h*)(sW3 + (kt * 32 + lane) * 16);
        acc3 = __builtin_amdgcn_wmma_f32_16x16x32_f16(
            false, a, false, b, (short)0, acc3, false, false);
    }

    // sigmoid + scatter color (divergence is fine: all WMMAs are done)
    if (n0 < 3) {
        float bb = b3[n0];
#pragma unroll
        for (int v = 0; v < 8; ++v)
            color[(base + mrow + v) * 3 + n0] = sigmoidf_dev(acc3[v] + bb);
    }
}

// ---------------------------------------------------------------------------
extern "C" void kernel_launch(void* const* d_in, const int* in_sizes, int n_in,
                              void* d_out, int out_size, void* d_ws, size_t ws_size,
                              hipStream_t stream) {
    const float* xyz   = (const float*)d_in[0];
    const float* grids = (const float*)d_in[1];
    const float* dens  = (const float*)d_in[2];
    const float* W1    = (const float*)d_in[3];
    const float* b1    = (const float*)d_in[4];
    const float* W2    = (const float*)d_in[5];
    const float* b2    = (const float*)d_in[6];
    const float* W3    = (const float*)d_in[7];
    const float* b3    = (const float*)d_in[8];

    float* out_density = (float*)d_out;
    float* out_color   = (float*)d_out + DENSITY_N;
    _Float16* pk       = (_Float16*)d_ws;   // 75776 B packed weights

    // density sigmoid: 2M floats as 524288 float4s
    density_sigmoid_kernel<<<DENSITY_N / 4 / 256, 256, 0, stream>>>(dens, out_density);

    // pack weights into WMMA B layout (f16)
    pack_weights_kernel<<<(74 * 32 + 255) / 256, 256, 0, stream>>>(W1, W2, W3, pk);

    // fused hash-encode + MLP: 2048 blocks x 8 waves x 16 points
    ngp_mlp_kernel<<<NPTS / 128, 256, 0, stream>>>(xyz, grids, pk,
                                                   b1, b2, b3, out_color);
}
